// ODEBlock_171798692376
// MI455X (gfx1250) — matrix operand order repaired and checked by reading, not compiled, not run on hardware
//
#include <hip/hip_runtime.h>
#include <math.h>

typedef __attribute__((ext_vector_type(16))) _Float16 v16h;
typedef __attribute__((ext_vector_type(8)))  _Float16 v8h;
typedef __attribute__((ext_vector_type(8)))  float    v8f;

#define FBB 32          // batch
#define HH 32
#define WW 32
#define CC 128          // channels
#define CI_PAD 160      // per-tap padded K (129 real: 128 data + 1 time)
#define KCHUNK 45       // 9 taps * 5 chunks of 32
#define APACK_ELEMS (KCHUNK * 8 * 32 * 16)     // halfs
#define NELEM (FBB * HH * WW * CC)             // 4194304 floats
#define N4 (NELEM / 4)
#define NB4 ((N4 + 255) / 256)
#define FLD_BYTES ((size_t)NELEM * 4)
#define OS_STRIDE 132   // padded f32 row stride for output transpose (bank spread)

// ---------------------------------------------------------------------------
// Weight packing: HWIO fp32 w[3][3][129][128] -> A-fragment-ready f16 layout
// Apack[kt][m][lane][j], lane holds row co = m*16 + (lane&15),
// chunk-local k = (j<8) ? ((lane>>4)*8 + j) : (16 + (lane>>4)*8 + j-8)
// ---------------------------------------------------------------------------
__global__ __launch_bounds__(256) void pack_weights(const float* __restrict__ w,
                                                    _Float16* __restrict__ Apack) {
    int idx = blockIdx.x * 256 + threadIdx.x;
    if (idx >= APACK_ELEMS) return;
    int j    = idx & 15;
    int lane = (idx >> 4) & 31;
    int m    = (idx >> 9) & 7;
    int kt   = idx >> 12;
    int tap  = kt / 5;
    int cib  = (kt - tap * 5) * 32;
    int kbA  = (lane >> 4) * 8;
    int klo  = (j < 8) ? (kbA + j) : (16 + kbA + (j - 8));
    int ci   = cib + klo;
    int co   = m * 16 + (lane & 15);
    float v  = 0.f;
    if (ci <= 128) v = w[((size_t)tap * 129 + ci) * 128 + co];
    Apack[idx] = (_Float16)v;
}

// ---------------------------------------------------------------------------
// Fused conv3x3(SAME, cat(x,t)) + bias + relu as implicit GEMM with WMMA f16.
// One workgroup per (b,h) row; 8 waves; wave i -> co tile i, both w tiles.
// Epilogue transposes through LDS for coalesced b128 output stores.
// ---------------------------------------------------------------------------
__global__ __launch_bounds__(256) void conv3x3_wmma(const float* __restrict__ xin,
                                                    const _Float16* __restrict__ Apack,
                                                    const float* __restrict__ bias,
                                                    float* __restrict__ out,
                                                    const float* __restrict__ sc,
                                                    const int* __restrict__ fl,
                                                    float ccoef) {
    if (fl[0]) return;                        // ODE integration already finished
    const float tst = sc[0] + ccoef * sc[2];  // stage time t + c*dt_c

    const int b = blockIdx.x >> 5;
    const int h = blockIdx.x & 31;

    // smem: f16 input stage [3][34][160] (32640 B), later reused as f32 [32][132]
    __shared__ __align__(16) char smem[3 * 34 * CI_PAD * 2];
    _Float16* Xs = (_Float16*)smem;

    // Vectorized staging: 8-channel groups; halo cols -1..32, ch128 = t, pad 0
    for (int g = threadIdx.x; g < 3 * 34 * 20; g += 256) {
        int r   = g / (34 * 20);
        int rem = g - r * (34 * 20);
        int col = rem / 20;
        int cg  = rem - col * 20;          // ci base = cg*8
        int gh  = h - 1 + r;
        int gw  = col - 1;
        v8h hv = {};
        if ((unsigned)gh < 32u && (unsigned)gw < 32u) {
            if (cg < 16) {
                const float4* p = (const float4*)(xin +
                    (((size_t)(b * 32 + gh) * 32 + gw) << 7) + cg * 8);
                float4 a = p[0], c = p[1];
                hv[0] = (_Float16)a.x; hv[1] = (_Float16)a.y;
                hv[2] = (_Float16)a.z; hv[3] = (_Float16)a.w;
                hv[4] = (_Float16)c.x; hv[5] = (_Float16)c.y;
                hv[6] = (_Float16)c.z; hv[7] = (_Float16)c.w;
            } else if (cg == 16) {
                hv[0] = (_Float16)tst;
            }
        }
        *(v8h*)(&Xs[(r * 34 + col) * CI_PAD + cg * 8]) = hv;
    }
    __syncthreads();

    const int wave = threadIdx.x >> 5;
    const int lane = threadIdx.x & 31;
    const int nl   = lane & 15;
    const int hi16 = lane >> 4;

    v8f acc0 = {};   // w tile 0 (cols 0..15)
    v8f acc1 = {};   // w tile 1 (cols 16..31)

    const _Float16* ap = Apack + wave * 512 + lane * 16;   // + kt*4096 per chunk

#pragma unroll 1
    for (int tap = 0; tap < 9; ++tap) {
        const int th = tap / 3, tw = tap % 3;
        const int rb0 = (th * 34 + nl + tw) * CI_PAD + hi16 * 8;    // col = w + tw
        const int rb1 = rb0 + 16 * CI_PAD;
        const _Float16* apt = ap + tap * 5 * 4096;
#pragma unroll
        for (int c5 = 0; c5 < 5; ++c5) {
            const int cib = c5 * 32;
            v16h a   = *(const v16h*)(apt + c5 * 4096);
            v16h bf0 = __builtin_shufflevector(*(const v8h*)(&Xs[rb0 + cib]), *(const v8h*)(&Xs[rb0 + cib + 16]), 0, 1, 2, 3, 4, 5, 6, 7, 8, 9, 10, 11, 12, 13, 14, 15);
            v16h bf1 = __builtin_shufflevector(*(const v8h*)(&Xs[rb1 + cib]), *(const v8h*)(&Xs[rb1 + cib + 16]), 0, 1, 2, 3, 4, 5, 6, 7, 8, 9, 10, 11, 12, 13, 14, 15);
            acc0 = __builtin_amdgcn_wmma_f32_16x16x32_f16(false, a, false, bf0,
                                                          (short)0, acc0, false, false);
            acc1 = __builtin_amdgcn_wmma_f32_16x16x32_f16(false, a, false, bf1,
                                                          (short)0, acc1, false, false);
        }
    }

    // D layout: lane L -> N = L&15 ; VGPR v -> M = v + 8*(L>>4); co contiguous
    __syncthreads();                 // all Xs reads done; reuse smem as f32 [32][132]
    float* Os = (float*)smem;
    const int co_base = wave * 16 + hi16 * 8;
    {
        const float4* bp = (const float4*)(bias + co_base);
        float4 b0 = bp[0], b1 = bp[1];
        float4 r00 = { fmaxf(acc0[0] + b0.x, 0.f), fmaxf(acc0[1] + b0.y, 0.f),
                       fmaxf(acc0[2] + b0.z, 0.f), fmaxf(acc0[3] + b0.w, 0.f) };
        float4 r01 = { fmaxf(acc0[4] + b1.x, 0.f), fmaxf(acc0[5] + b1.y, 0.f),
                       fmaxf(acc0[6] + b1.z, 0.f), fmaxf(acc0[7] + b1.w, 0.f) };
        float4 r10 = { fmaxf(acc1[0] + b0.x, 0.f), fmaxf(acc1[1] + b0.y, 0.f),
                       fmaxf(acc1[2] + b0.z, 0.f), fmaxf(acc1[3] + b0.w, 0.f) };
        float4 r11 = { fmaxf(acc1[4] + b1.x, 0.f), fmaxf(acc1[5] + b1.y, 0.f),
                       fmaxf(acc1[6] + b1.z, 0.f), fmaxf(acc1[7] + b1.w, 0.f) };
        *(float4*)&Os[nl * OS_STRIDE + co_base]            = r00;
        *(float4*)&Os[nl * OS_STRIDE + co_base + 4]        = r01;
        *(float4*)&Os[(16 + nl) * OS_STRIDE + co_base]     = r10;
        *(float4*)&Os[(16 + nl) * OS_STRIDE + co_base + 4] = r11;
    }
    __syncthreads();

    // Coalesced b128 writeback of the 32x128 f32 row
    float4* outv = (float4*)(out + (size_t)blockIdx.x * 32 * 128);
    for (int i = threadIdx.x; i < 1024; i += 256) {
        int row = i >> 5, c4 = i & 31;
        outv[i] = *(const float4*)&Os[row * OS_STRIDE + (c4 << 2)];
    }
}

// ---------------------------------------------------------------------------
// out = y + dt_c * (c1*k1 + c2*k2 + c3*k3 + c4*k4 + c5*k5)   (float4 vectorized)
// ---------------------------------------------------------------------------
__global__ __launch_bounds__(256) void axpy5(const float4* __restrict__ y,
                                             const float4* __restrict__ k1,
                                             const float4* __restrict__ k2,
                                             const float4* __restrict__ k3,
                                             const float4* __restrict__ k4,
                                             const float4* __restrict__ k5,
                                             float c1, float c2, float c3, float c4, float c5,
                                             const float* __restrict__ sc,
                                             const int* __restrict__ fl,
                                             float4* __restrict__ out) {
    if (fl[0]) return;
    int i = blockIdx.x * 256 + threadIdx.x;
    if (i >= N4) return;
    const float dt = sc[2];
    float4 Y = y[i], A = k1[i], B = k2[i], C = k3[i], D = k4[i], E = k5[i];
    float4 o;
    o.x = Y.x + dt * (c1 * A.x + c2 * B.x + c3 * C.x + c4 * D.x + c5 * E.x);
    o.y = Y.y + dt * (c1 * A.y + c2 * B.y + c3 * C.y + c4 * D.y + c5 * E.y);
    o.z = Y.z + dt * (c1 * A.z + c2 * B.z + c3 * C.z + c4 * D.z + c5 * E.z);
    o.w = Y.w + dt * (c1 * A.w + c2 * B.w + c3 * C.w + c4 * D.w + c5 * E.w);
    out[i] = o;
}

// ---------------------------------------------------------------------------
// Fused error estimate + squared-scaled-norm partial reduction into sc[3]
// ---------------------------------------------------------------------------
__global__ __launch_bounds__(256) void err_reduce(const float4* __restrict__ y,
                                                  const float4* __restrict__ y5,
                                                  const float4* __restrict__ k1,
                                                  const float4* __restrict__ k3,
                                                  const float4* __restrict__ k4,
                                                  const float4* __restrict__ k5,
                                                  const float4* __restrict__ k6,
                                                  const float4* __restrict__ k7,
                                                  float* __restrict__ sc,
                                                  const int* __restrict__ fl) {
    if (fl[0]) return;
    const float dt = sc[2];
    const float e1 = 35.f / 384.f - 5179.f / 57600.f;
    const float e3 = 500.f / 1113.f - 7571.f / 16695.f;
    const float e4 = 125.f / 192.f - 393.f / 640.f;
    const float e5 = -2187.f / 6784.f + 92097.f / 339200.f;
    const float e6 = 11.f / 84.f - 187.f / 2100.f;
    const float e7 = -1.f / 40.f;

    float s = 0.f;
    int i = blockIdx.x * 256 + threadIdx.x;
    if (i < N4) {
        float4 Y = y[i], Y5 = y5[i];
        float4 K1 = k1[i], K3 = k3[i], K4 = k4[i], K5 = k5[i], K6 = k6[i], K7 = k7[i];
        {
            float e = dt * (e1 * K1.x + e3 * K3.x + e4 * K4.x + e5 * K5.x + e6 * K6.x + e7 * K7.x);
            float scale = 1e-3f + 1e-3f * fmaxf(fabsf(Y.x), fabsf(Y5.x));
            float r = e / scale; s += r * r;
        }
        {
            float e = dt * (e1 * K1.y + e3 * K3.y + e4 * K4.y + e5 * K5.y + e6 * K6.y + e7 * K7.y);
            float scale = 1e-3f + 1e-3f * fmaxf(fabsf(Y.y), fabsf(Y5.y));
            float r = e / scale; s += r * r;
        }
        {
            float e = dt * (e1 * K1.z + e3 * K3.z + e4 * K4.z + e5 * K5.z + e6 * K6.z + e7 * K7.z);
            float scale = 1e-3f + 1e-3f * fmaxf(fabsf(Y.z), fabsf(Y5.z));
            float r = e / scale; s += r * r;
        }
        {
            float e = dt * (e1 * K1.w + e3 * K3.w + e4 * K4.w + e5 * K5.w + e6 * K6.w + e7 * K7.w);
            float scale = 1e-3f + 1e-3f * fmaxf(fabsf(Y.w), fabsf(Y5.w));
            float r = e / scale; s += r * r;
        }
    }
    __shared__ float red[256];
    red[threadIdx.x] = s;
    __syncthreads();
    for (int off = 128; off > 0; off >>= 1) {
        if (threadIdx.x < off) red[threadIdx.x] += red[threadIdx.x + off];
        __syncthreads();
    }
    if (threadIdx.x == 0) atomicAdd(&sc[3], red[0]);
}

// ---------------------------------------------------------------------------
// Control kernels (1 thread) implementing the adaptive dopri5 controller
// sc: [0]=t [1]=dt [2]=dt_c [3]=err_sum   fl: [0]=done [1]=accept(adv)
// ---------------------------------------------------------------------------
__global__ void init_sc(float* sc, int* fl) {
    sc[0] = 0.f; sc[1] = 0.05f; sc[2] = 0.05f; sc[3] = 0.f;
    fl[0] = 0; fl[1] = 0;
}

__global__ void ctrl_pre(float* sc, int* fl) {
    if (fl[0]) return;
    float t = sc[0], dt = sc[1];
    sc[2] = fminf(dt, 1.0f - t);
    sc[3] = 0.f;
    fl[1] = 0;
}

__global__ void ctrl_post(float* sc, int* fl) {
    if (fl[0]) { fl[1] = 0; return; }
    float t = sc[0], dtc = sc[2];
    float en = sqrtf(sc[3] / (float)NELEM);
    en = fmaxf(en, 1e-10f);
    int accept = (en <= 1.0f) ? 1 : 0;
    float factor = 0.9f * powf(en, -0.2f);
    factor = fminf(fmaxf(factor, 0.2f), 10.f);
    if (accept) { t += dtc; sc[0] = t; }
    sc[1] = dtc * factor;
    fl[1] = accept;
    if (t >= 1.0f - 1e-6f) fl[0] = 1;
}

__global__ __launch_bounds__(256) void commit(float4* __restrict__ y,
                                              const float4* __restrict__ y5,
                                              const int* __restrict__ fl) {
    if (!fl[1]) return;
    int i = blockIdx.x * 256 + threadIdx.x;
    if (i < N4) y[i] = y5[i];
}

// ---------------------------------------------------------------------------
extern "C" void kernel_launch(void* const* d_in, const int* in_sizes, int n_in,
                              void* d_out, int out_size, void* d_ws, size_t ws_size,
                              hipStream_t stream) {
    (void)in_sizes; (void)n_in; (void)out_size; (void)ws_size;
    const float* x  = (const float*)d_in[0];
    const float* w1 = (const float*)d_in[1];
    const float* b1 = (const float*)d_in[2];
    const float* w2 = (const float*)d_in[3];
    const float* b2 = (const float*)d_in[4];
    float* y = (float*)d_out;

    char* ws = (char*)d_ws;
    float* sc = (float*)ws;
    int*   fl = (int*)(ws + 64);
    _Float16* Ap1 = (_Float16*)(ws + 4096);
    _Float16* Ap2 = (_Float16*)(ws + 4096 + (1 << 19));
    char* fields = ws + (2u << 20);
    float* hbuf  = (float*)(fields + 0 * FLD_BYTES);
    float* ytmp  = (float*)(fields + 1 * FLD_BYTES);
    float* y5b   = (float*)(fields + 2 * FLD_BYTES);
    float* k1    = (float*)(fields + 3 * FLD_BYTES);
    float* k2    = (float*)(fields + 4 * FLD_BYTES);
    float* k3    = (float*)(fields + 5 * FLD_BYTES);
    float* k4    = (float*)(fields + 6 * FLD_BYTES);
    float* k5    = (float*)(fields + 7 * FLD_BYTES);
    float* k6    = (float*)(fields + 8 * FLD_BYTES);
    float* k7    = (float*)(fields + 9 * FLD_BYTES);

    const int packBlocks = (APACK_ELEMS + 255) / 256;
    pack_weights<<<packBlocks, 256, 0, stream>>>(w1, Ap1);
    pack_weights<<<packBlocks, 256, 0, stream>>>(w2, Ap2);
    init_sc<<<1, 1, 0, stream>>>(sc, fl);
    hipMemcpyAsync(y, x, FLD_BYTES, hipMemcpyDeviceToDevice, stream);

    auto F4 = [](float* p) { return (float4*)p; };
    auto feval = [&](const float* yin, float cc, float* kout) {
        conv3x3_wmma<<<FBB * HH, 256, 0, stream>>>(yin, Ap1, b1, hbuf, sc, fl, cc);
        conv3x3_wmma<<<FBB * HH, 256, 0, stream>>>(hbuf, Ap2, b2, kout, sc, fl, cc);
    };
    auto comb = [&](float* out, const float* ka, const float* kb, const float* kc,
                    const float* kd, const float* ke,
                    float c1, float c2, float c3, float c4, float c5) {
        axpy5<<<NB4, 256, 0, stream>>>((const float4*)y, (const float4*)ka, (const float4*)kb,
                                       (const float4*)kc, (const float4*)kd, (const float4*)ke,
                                       c1, c2, c3, c4, c5, sc, fl, F4(out));
    };

    for (int step = 0; step < 32; ++step) {
        ctrl_pre<<<1, 1, 0, stream>>>(sc, fl);

        feval(y, 0.f, k1);
        comb(ytmp, k1, k1, k1, k1, k1, 0.2f, 0.f, 0.f, 0.f, 0.f);
        feval(ytmp, 0.2f, k2);
        comb(ytmp, k1, k2, k1, k1, k1, 3.f / 40.f, 9.f / 40.f, 0.f, 0.f, 0.f);
        feval(ytmp, 0.3f, k3);
        comb(ytmp, k1, k2, k3, k1, k1, 44.f / 45.f, -56.f / 15.f, 32.f / 9.f, 0.f, 0.f);
        feval(ytmp, 0.8f, k4);
        comb(ytmp, k1, k2, k3, k4, k1,
             19372.f / 6561.f, -25360.f / 2187.f, 64448.f / 6561.f, -212.f / 729.f, 0.f);
        feval(ytmp, 8.f / 9.f, k5);
        comb(ytmp, k1, k2, k3, k4, k5,
             9017.f / 3168.f, -355.f / 33.f, 46732.f / 5247.f, 49.f / 176.f, -5103.f / 18656.f);
        feval(ytmp, 1.f, k6);
        comb(y5b, k1, k3, k4, k5, k6,
             35.f / 384.f, 500.f / 1113.f, 125.f / 192.f, -2187.f / 6784.f, 11.f / 84.f);
        feval(y5b, 1.f, k7);   // FSAL stage for the error estimate

        err_reduce<<<NB4, 256, 0, stream>>>((const float4*)y, (const float4*)y5b,
                                            (const float4*)k1, (const float4*)k3,
                                            (const float4*)k4, (const float4*)k5,
                                            (const float4*)k6, (const float4*)k7, sc, fl);
        ctrl_post<<<1, 1, 0, stream>>>(sc, fl);
        commit<<<NB4, 256, 0, stream>>>(F4(y), (const float4*)y5b, fl);
    }
}
